// Selector_39685497815886
// MI455X (gfx1250) — matrix operand order repaired and checked by reading, not compiled, not run on hardware
//
#include <hip/hip_runtime.h>
#include <hip/hip_bf16.h>
#include <cstdint>
#include <cstddef>

// ---------------- types ----------------
typedef __bf16 bf16_t;
typedef __attribute__((ext_vector_type(16))) __bf16 v16bf;
typedef __attribute__((ext_vector_type(8)))  __bf16 v8bf;
typedef __attribute__((ext_vector_type(8)))  float  v8f;
typedef __attribute__((ext_vector_type(4)))  float  v4f;

typedef int v4i __attribute__((vector_size(16)));
typedef __attribute__((address_space(1))) v4i as1_v4i;
typedef __attribute__((address_space(3))) v4i as3_v4i;

// ---------------- problem constants ----------------
#define PP      128     // num_shards_prefetched
#define SS      8       // num_shards_selected
#define BB      4
#define TT      4096
#define EE      1024
#define TTILE   128     // t-tile per workgroup (16 t per wave x 8 waves)
#define KSTEP   32      // bf16 WMMA K
#define NKS     (EE / KSTEP)
#define LDSROW  36      // padded LDS row stride in floats (32 data + 4 pad)

#if __has_builtin(__builtin_amdgcn_global_load_async_to_lds_b128)
#define HAVE_ASYNC_LDS 1
#else
#define HAVE_ASYNC_LDS 0
#endif

template <int N>
__device__ __forceinline__ void wait_async() {
#if HAVE_ASYNC_LDS
#if __has_builtin(__builtin_amdgcn_s_wait_asynccnt)
  __builtin_amdgcn_s_wait_asynccnt(N);
#else
  asm volatile("s_wait_asynccnt %0" ::"n"(N) : "memory");
#endif
#endif
}

// Stage one wave-private tile: 16 rows x 32 f32 (global row stride EE) into LDS
// (row stride LDSROW). 128 chunks of 16B, 4 per lane.
__device__ __forceinline__ void stage_tile(const float* g_rows, float* l_rows, int lane) {
#pragma unroll
  for (int c = 0; c < 4; ++c) {
    const int idx = lane * 4 + c;       // 0..127
    const int row = idx >> 3;           // 8 chunks per 32-float row
    const int col = (idx & 7) << 2;     // float offset within row
    const float* g = g_rows + (size_t)row * EE + col;
    float* l = l_rows + row * LDSROW + col;
#if HAVE_ASYNC_LDS
    __builtin_amdgcn_global_load_async_to_lds_b128((as1_v4i*)g, (as3_v4i*)l, 0, 0);
#else
    v4f v = *(const v4f*)g;
    *(v4f*)l = v;
#endif
  }
}

// ---------------- kernel 0: split keys f32 -> bf16 hi/lo, zero loss accum ----------------
__global__ void selector_prep(const float* __restrict__ keys,
                              bf16_t* __restrict__ khi, bf16_t* __restrict__ klo,
                              float* __restrict__ lossbuf) {
  const int i = blockIdx.x * blockDim.x + threadIdx.x;
  if (i < 2) lossbuf[i] = 0.0f;
  if (i < PP * EE) {
    const float x = keys[i];
    const __bf16 h = (__bf16)x;
    khi[i] = h;
    klo[i] = (__bf16)(x - (float)h);
  }
}

// ---------------- kernel 1: fused GEMM (bf16x3 WMMA) + topk + softmax + loss partials ----
__global__ __launch_bounds__(256) void selector_main(
    const float* __restrict__ tensor, const bf16_t* __restrict__ khi,
    const bf16_t* __restrict__ klo, const float* __restrict__ biases,
    float* __restrict__ out, float* __restrict__ lossbuf) {
  __shared__ float lds[2][8][16 * LDSROW];   // double-buffered, per-wave private slabs

  const int wave = threadIdx.x >> 5;
  const int lane = threadIdx.x & 31;
  const int lh   = lane & 15;
  const int hi   = lane >> 4;                 // lane half: selects K/M sub-chunks

  const int b     = blockIdx.x >> 5;          // 32 t-tiles per batch
  const int ttile = blockIdx.x & 31;
  const int tw    = ttile * TTILE + wave * 16;  // wave's first t
  const int t     = tw + lh;                    // this lane's t (N index)

  const float* gw = tensor + (size_t)(b * TT + tw) * EE;

  // C tile: M=p (128, 8 sub-tiles of 16), N=t (16). Lane holds, per p-tile q,
  // p = q*16 + hi*8 + m in acc[q][m]  (ISA 16x16 f32 C/D layout).
  v8f acc[8];
  const v8f vzero = {0.f, 0.f, 0.f, 0.f, 0.f, 0.f, 0.f, 0.f};
#pragma unroll
  for (int q = 0; q < 8; ++q) acc[q] = vzero;

  stage_tile(gw, &lds[0][wave][0], lane);

#pragma unroll 1
  for (int ks = 0; ks < NKS; ++ks) {
    const int cur = ks & 1;
    if (ks + 1 < NKS) {
      stage_tile(gw + (ks + 1) * KSTEP, &lds[cur ^ 1][wave][0], lane);
      wait_async<4>();   // current buffer's 4 chunks done; next tile in flight
    } else {
      wait_async<0>();
    }

    // B fragment (32x16 bf16): lane holds column t, K = k0 + hi*8 + [0..7] and k0 + 16 + hi*8 + [0..7],
    // converted f32 -> bf16 hi/lo on the fly.
    const float* rp = &lds[cur][wave][lh * LDSROW + hi * 8];
    v16bf bh, bl;
#pragma unroll
    for (int j = 0; j < 16; ++j) {
      const float x = rp[(j & 7) + ((j >> 3) << 4)];
      const __bf16 h = (__bf16)x;
      bh[j] = h;
      bl[j] = (__bf16)(x - (float)h);
    }

    const int k0 = ks * KSTEP;
    const int koffA = hi * 8;   // A 16x32 bf16 layout: lanes 0-15 K 0..7/16..23, 16-31 K 8..15/24..31
#pragma unroll
    for (int q = 0; q < 8; ++q) {
      const size_t abase = (size_t)(q * 16 + lh) * EE + k0 + koffA;
      const v8bf h0 = *(const v8bf*)(khi + abase);
      const v8bf h1 = *(const v8bf*)(khi + abase + 16);
      const v8bf l0 = *(const v8bf*)(klo + abase);
      const v8bf l1 = *(const v8bf*)(klo + abase + 16);
      const v16bf ah = __builtin_shufflevector(h0, h1, 0,1,2,3,4,5,6,7,8,9,10,11,12,13,14,15);
      const v16bf al = __builtin_shufflevector(l0, l1, 0,1,2,3,4,5,6,7,8,9,10,11,12,13,14,15);
      // fp32-emulating bf16x3: hi*hi + hi*lo + lo*hi
      acc[q] = __builtin_amdgcn_wmma_f32_16x16x32_bf16(false, ah, false, bh, (short)0, acc[q], false, false);
      acc[q] = __builtin_amdgcn_wmma_f32_16x16x32_bf16(false, ah, false, bl, (short)0, acc[q], false, false);
      acc[q] = __builtin_amdgcn_wmma_f32_16x16x32_bf16(false, al, false, bh, (short)0, acc[q], false, false);
    }
  }

  // ---- loss partials from raw scores (pre-bias): per-t mean/std over the 128 p ----
  float sum = 0.f, sq = 0.f;
#pragma unroll
  for (int q = 0; q < 8; ++q)
#pragma unroll
    for (int m = 0; m < 8; ++m) { const float v = acc[q][m]; sum += v; sq += v * v; }
  sum += __shfl_xor(sum, 16);   // partner lane holds the other 64 p for the same t
  sq  += __shfl_xor(sq, 16);
  const float mean = sum * (1.f / 128.f);
  float var = (sq - 128.f * mean * mean) * (1.f / 127.f);   // ddof=1
  var = var > 0.f ? var : 0.f;
  const float ds = 1.f - sqrtf(var);
  float l1 = mean * mean;
  float l2 = ds * ds;
#pragma unroll
  for (int off = 16; off > 0; off >>= 1) {
    l1 += __shfl_xor(l1, off);
    l2 += __shfl_xor(l2, off);
  }
  if (lane == 0) {              // each t counted twice in the wave sum -> 0.5x
    atomicAdd(&lossbuf[0], 0.5f * l1);
    atomicAdd(&lossbuf[1], 0.5f * l2);
  }

  // ---- top-8 of 128 biased scores, per lane (ascending p => stable ties) ----
  float tv[SS]; int ti[SS];
#pragma unroll
  for (int s = 0; s < SS; ++s) { tv[s] = -3.4e38f; ti[s] = 0; }

#pragma unroll
  for (int q = 0; q < 8; ++q) {
    float pv[8];
#pragma unroll
    for (int m = 0; m < 8; ++m) pv[m] = __shfl_xor(acc[q][m], 16);
#pragma unroll
    for (int mm = 0; mm < 16; ++mm) {
      const int m = mm & 7;
      // own regs hold p = q*16 + hi*8 + m; partner's hold the other half
      float v = ((mm >= 8) == (hi != 0)) ? acc[q][m] : pv[m];
      const int p = q * 16 + mm;
      v += biases[p];
      if (v > tv[SS - 1]) {
        tv[SS - 1] = v; ti[SS - 1] = p;
#pragma unroll
        for (int j = SS - 1; j > 0; --j) {
          if (tv[j] > tv[j - 1]) {
            const float tf = tv[j]; tv[j] = tv[j - 1]; tv[j - 1] = tf;
            const int tx = ti[j]; ti[j] = ti[j - 1]; ti[j - 1] = tx;
          }
        }
      }
    }
  }

  // ---- softmax over the 8 selected; write selections + weights (lane pair -> once) ----
  if (hi == 0) {
    const float mx = tv[0];
    float e[SS], tot = 0.f;
#pragma unroll
    for (int s = 0; s < SS; ++s) { e[s] = __expf(tv[s] - mx); tot += e[s]; }
    const float inv = 1.f / tot;
    const size_t selbase = (size_t)b * SS * TT + t;
    float* wout = out + (size_t)BB * SS * TT;
#pragma unroll
    for (int s = 0; s < SS; ++s) {
      out[selbase + (size_t)s * TT]  = (float)ti[s];
      wout[selbase + (size_t)s * TT] = e[s] * inv;
    }
  }
}

// ---------------- kernel 2: finalize scalar loss ----------------
// (the connectome/nudge term is exactly zero in value; gradient-only)
__global__ void selector_finalize(const float* __restrict__ lossbuf, float* __restrict__ out) {
  if (threadIdx.x == 0 && blockIdx.x == 0) {
    const float invBT = 1.f / (float)(BB * TT);
    out[(size_t)2 * BB * SS * TT] = 0.01f * lossbuf[0] * invBT + 0.01f * lossbuf[1] * invBT;
  }
}

// ---------------- launch ----------------
extern "C" void kernel_launch(void* const* d_in, const int* in_sizes, int n_in,
                              void* d_out, int out_size, void* d_ws, size_t ws_size,
                              hipStream_t stream) {
  const float* tensor = (const float*)d_in[0];   // (4, 4096, 1024) f32
  const float* keys   = (const float*)d_in[1];   // (128, 1024) f32
  const float* biases = (const float*)d_in[2];   // (128,) f32
  // d_in[3] partitions / d_in[4] connectome_biases: loss contribution is exactly 0

  bf16_t* khi = (bf16_t*)d_ws;                               // 128*1024 bf16
  bf16_t* klo = khi + (size_t)PP * EE;                       // 128*1024 bf16
  float* lossbuf = (float*)((char*)d_ws + (size_t)2 * PP * EE * sizeof(bf16_t));
  float* out = (float*)d_out;

  selector_prep<<<(PP * EE + 255) / 256, 256, 0, stream>>>(keys, khi, klo, lossbuf);
  selector_main<<<BB * (TT / TTILE), 256, 0, stream>>>(tensor, khi, klo, biases, out, lossbuf);
  selector_finalize<<<1, 1, 0, stream>>>(lossbuf, out);
}
